// MoETransformer_80204219285857
// MI455X (gfx1250) — matrix operand order repaired and checked by reading, not compiled, not run on hardware
//
#include <hip/hip_runtime.h>
#include <hip/hip_bf16.h>

#define DMODEL 768
#define SEQ    1024
#define NBATCH 4
#define NTOK   4096      // B*S
#define NHEADS 12
#define DHEADC 64
#define NEXP   8
#define DHID   1536
#define VOCAB  512
#define NLAYER 4

typedef __attribute__((ext_vector_type(16))) __bf16 v16bf;
typedef __attribute__((ext_vector_type(8)))  __bf16 v8bf;
typedef __attribute__((ext_vector_type(8)))  float  v8f;
typedef __attribute__((ext_vector_type(4)))  float  v4f;

__device__ inline v8f wmma_bf16(v16bf a, v16bf b, v8f c) {
  return __builtin_amdgcn_wmma_f32_16x16x32_bf16(false, a, false, b, (short)0, c,
                                                 false, false);
}

// Two aligned 16B LDS reads -> one 16-element bf16 fragment (ds_load_b128 x2).
__device__ inline v16bf ld16(const __bf16* p0, const __bf16* p1) {
  v8bf lo = *(const v8bf*)p0;
  v8bf hi = *(const v8bf*)p1;
  return __builtin_shufflevector(lo, hi, 0, 1, 2, 3, 4, 5, 6, 7, 8, 9, 10, 11,
                                 12, 13, 14, 15);
}

__device__ inline v8bf cvt8(v4f f0, v4f f1) {
  v8bf r;
#pragma unroll
  for (int j = 0; j < 4; ++j) { r[j] = (__bf16)f0[j]; r[4 + j] = (__bf16)f1[j]; }
  return r;
}

// CDNA5 async copy: 16B global -> LDS without touching VGPR data path.
__device__ inline void async_to_lds_b128(const void* gsrc, void* ldsDst) {
  unsigned off = (unsigned)(uintptr_t)ldsDst;  // low 32b of generic = LDS offset
  asm volatile("global_load_async_to_lds_b128 %0, %1, off"
               :
               : "v"(off), "v"(gsrc)
               : "memory");
}
__device__ inline void wait_asynccnt0() {
  asm volatile("s_wait_asynccnt 0x0" ::: "memory");
}

__device__ inline float gelu_tanh(float x) {
  const float k0 = 0.7978845608028654f, k1 = 0.044715f;
  return 0.5f * x * (1.f + tanhf(k0 * (x + k1 * x * x * x)));
}

// ---------------------------------------------------------------- embedding
__global__ __launch_bounds__(256) void embed_kernel(
    const int* __restrict__ x, const float* __restrict__ tok,
    const float* __restrict__ pos, float* __restrict__ h) {
  int i = blockIdx.x * 256 + threadIdx.x;     // i < NTOK*DMODEL exactly
  int n = i / DMODEL, d = i - n * DMODEL;
  int t = x[n];
  int s = n % SEQ;
  h[i] = tok[(size_t)t * DMODEL + d] + pos[(size_t)s * DMODEL + d];
}

// ---------------------------------------------------------------- layernorm
__global__ __launch_bounds__(256) void layernorm_kernel(
    const float* __restrict__ x, const float* __restrict__ g,
    const float* __restrict__ b, __bf16* __restrict__ out) {
  __shared__ float r1[256], r2[256];
  int n = blockIdx.x, tid = threadIdx.x;
  const float* xr = x + (size_t)n * DMODEL;
  float s = 0.f, s2 = 0.f;
  for (int d = tid; d < DMODEL; d += 256) { float v = xr[d]; s += v; s2 += v * v; }
  r1[tid] = s; r2[tid] = s2; __syncthreads();
  for (int off = 128; off > 0; off >>= 1) {
    if (tid < off) { r1[tid] += r1[tid + off]; r2[tid] += r2[tid + off]; }
    __syncthreads();
  }
  float mean = r1[0] / DMODEL;
  float var  = r2[0] / DMODEL - mean * mean;
  float inv  = rsqrtf(var + 1e-5f);
  for (int d = tid; d < DMODEL; d += 256)
    out[(size_t)n * DMODEL + d] = (__bf16)((xr[d] - mean) * inv * g[d] + b[d]);
}

// ------------------------------------------------ generic bf16 WMMA GEMM
// C[M,N] = A[M,K](bf16, async-copied to LDS) * B[K,N](fp32 -> bf16, transposed)
// Double-buffered LDS, 2-stage software pipeline.
// EPI 0: Cf = v          (qkv / logits)
// EPI 1: Cf += v         (proj + residual)
// EPI 2: Cb = bf16(gelu(v))           (MoE w1)
// EPI 3: Cf[rowIdx[m]] += rowWgt[m]*v (MoE w2 weighted scatter)
#define BM 64
#define BN 64
#define BK 32

template <int EPI, bool BT>
__global__ __launch_bounds__(256) void gemm_bf16_kernel(
    const __bf16* __restrict__ A, const float* __restrict__ Bsrc,
    const float* __restrict__ bias, float* __restrict__ Cf,
    __bf16* __restrict__ Cb, const int* __restrict__ rowIdx,
    const float* __restrict__ rowWgt, const int* __restrict__ cntPtr,
    int M, int K, int N, int ldc) {
  __shared__ __align__(16) __bf16 As[2][BM][BK + 8];   // row-major [m][k]
  __shared__ __align__(16) __bf16 Bt[2][BN][BK + 8];   // transposed [n][k]

  if (cntPtr) M = cntPtr[0];
  int mBase = blockIdx.x * BM;
  if (mBase >= M) return;
  int nBase = blockIdx.y * BN;

  int tid = threadIdx.x;
  int lane = tid & 31;
  int w = tid >> 5;
  int mw = (w >> 2) * 32;   // 2 waves over M (32 rows each -> 2 wmma tiles)
  int nw = (w & 3) * 16;    // 4 waves over N (16 cols each)
  int lr = lane & 15;
  int hi = lane >> 4;

  // staging coordinates: A thread -> 8 contiguous bf16 of one row
  int ra = tid >> 2, ca = (tid & 3) * 8;
  int growA = mBase + ra;
  // B thread -> 8 K-values of one N column (lane-coalesced global reads)
  int nB = tid & 63, kc = (tid >> 6) * 8;

  auto stageA = [&](int ks, int buf) {
    __bf16* dst = &As[buf][ra][ca];
    if (growA < M) {
      async_to_lds_b128(A + (size_t)growA * K + ks * BK + ca, dst);
    } else {
      v8bf z;
#pragma unroll
      for (int j = 0; j < 8; ++j) z[j] = (__bf16)0.f;
      *(v8bf*)dst = z;
    }
  };
  auto stageB = [&](int ks, int buf) {
    int kBase = ks * BK;
    v8bf bv;
    if (!BT) {
      const float* src = Bsrc + (size_t)(kBase + kc) * N + nBase + nB;
#pragma unroll
      for (int j = 0; j < 8; ++j) bv[j] = (__bf16)src[(size_t)j * N];
    } else {
      const float* src = Bsrc + (size_t)(nBase + nB) * K + kBase + kc;
      v4f f0 = *(const v4f*)src, f1 = *(const v4f*)(src + 4);
      bv = cvt8(f0, f1);
    }
    *(v8bf*)&Bt[buf][nB][kc] = bv;  // one ds_store_b128
  };

  v8f acc0, acc1;
#pragma unroll
  for (int i = 0; i < 8; ++i) { acc0[i] = 0.f; acc1[i] = 0.f; }

  int ksteps = K / BK;
  stageA(0, 0);
  stageB(0, 0);
  wait_asynccnt0();
  __syncthreads();

  int cur = 0;
  for (int ks = 0; ks < ksteps; ++ks) {
    int nxt = cur ^ 1;
    if (ks + 1 < ksteps) {   // pipeline next tile while computing this one
      stageA(ks + 1, nxt);
      stageB(ks + 1, nxt);
      if (ks + 2 < ksteps) {  // L2 prefetch two tiles ahead for the fp32 stream
        if (!BT)
          __builtin_prefetch(Bsrc + (size_t)((ks + 2) * BK + kc) * N + nBase + nB, 0, 0);
        else
          __builtin_prefetch(Bsrc + (size_t)(nBase + nB) * K + (ks + 2) * BK + kc, 0, 0);
      }
    }

    const __bf16* ar0 = &As[cur][mw + lr][0];
    const __bf16* ar1 = &As[cur][mw + 16 + lr][0];
    const __bf16* br  = &Bt[cur][nw + lr][0];
    v16bf a0 = ld16(ar0 + hi * 8, ar0 + 16 + hi * 8);
    v16bf a1 = ld16(ar1 + hi * 8, ar1 + 16 + hi * 8);
    v16bf bf = ld16(br + hi * 8, br + 16 + hi * 8);
    acc0 = wmma_bf16(a0, bf, acc0);
    acc1 = wmma_bf16(a1, bf, acc1);

    wait_asynccnt0();   // next tile's async A writes have landed in LDS
    __syncthreads();    // everyone done reading cur / writing nxt
    cur = nxt;
  }

#pragma unroll
  for (int t = 0; t < 2; ++t) {
    v8f& acc = t ? acc1 : acc0;
#pragma unroll
    for (int r = 0; r < 8; ++r) {
      int row = mBase + mw + t * 16 + r + 8 * hi;
      if (row >= M) continue;
      int col = nBase + nw + lr;
      float val = acc[r] + (bias ? bias[col] : 0.f);
      if constexpr (EPI == 0) {
        Cf[(size_t)row * ldc + col] = val;
      } else if constexpr (EPI == 1) {
        Cf[(size_t)row * ldc + col] += val;
      } else if constexpr (EPI == 2) {
        Cb[(size_t)row * ldc + col] = (__bf16)gelu_tanh(val);
      } else {
        int n = rowIdx[row];
        Cf[(size_t)n * ldc + col] += rowWgt[row] * val;
      }
    }
  }
}

// ------------------------------------------- flash attention (WMMA QK^T & PV)
__global__ __launch_bounds__(256) void attn_kernel(
    const float* __restrict__ qkv, __bf16* __restrict__ out) {
  __shared__ __align__(16) __bf16 Qs[64][72];   // [q][d]
  __shared__ __align__(16) __bf16 Ks[64][72];   // [k][d] (B^T orientation)
  __shared__ __align__(16) __bf16 Vt[64][72];   // transposed: [d][k]
  __shared__ __align__(16) __bf16 Pb[64][72];   // [q][k]
  __shared__ float  Sc[64][65];
  __shared__ float rowM[64], rowL[64], rowScale[64];

  int qt = blockIdx.x, hh = blockIdx.y, bb = blockIdx.z;
  int tid = threadIdx.x, lane = tid & 31, w = tid >> 5;
  int lr = lane & 15, hi = lane >> 4;
  int qs = (w >> 1) * 16;   // wave's 16-row q strip
  int cs = (w & 1) * 32;    // wave's 32-col strip (2 tiles)
  int rstage = tid >> 2, cstage = (tid & 3) * 16;

  {  // load Q tile (fp32 -> bf16) with vector loads
    const float* src = qkv + (size_t)(bb * SEQ + qt * 64 + rstage) * (3 * DMODEL) +
                       hh * DHEADC + cstage;
    v4f f0 = *(const v4f*)src, f1 = *(const v4f*)(src + 4);
    v4f f2 = *(const v4f*)(src + 8), f3 = *(const v4f*)(src + 12);
    *(v8bf*)&Qs[rstage][cstage] = cvt8(f0, f1);
    *(v8bf*)&Qs[rstage][cstage + 8] = cvt8(f2, f3);
  }
  if (tid < 64) { rowM[tid] = -3.0e38f; rowL[tid] = 0.f; }

  v8f acc0, acc1;
#pragma unroll
  for (int i = 0; i < 8; ++i) { acc0[i] = 0.f; acc1[i] = 0.f; }

  for (int kt = 0; kt <= qt; ++kt) {
    __syncthreads();  // previous PV / Q-load done before overwriting K,V
    {
      const float* ksrc = qkv +
          (size_t)(bb * SEQ + kt * 64 + rstage) * (3 * DMODEL) + DMODEL +
          hh * DHEADC + cstage;
      const float* vsrc = ksrc + DMODEL;
      v4f f0 = *(const v4f*)ksrc, f1 = *(const v4f*)(ksrc + 4);
      v4f f2 = *(const v4f*)(ksrc + 8), f3 = *(const v4f*)(ksrc + 12);
      *(v8bf*)&Ks[rstage][cstage] = cvt8(f0, f1);
      *(v8bf*)&Ks[rstage][cstage + 8] = cvt8(f2, f3);
      v4f g0 = *(const v4f*)vsrc, g1 = *(const v4f*)(vsrc + 4);
      v4f g2 = *(const v4f*)(vsrc + 8), g3 = *(const v4f*)(vsrc + 12);
#pragma unroll
      for (int j = 0; j < 4; ++j) {
        Vt[cstage + j][rstage]      = (__bf16)g0[j];
        Vt[cstage + 4 + j][rstage]  = (__bf16)g1[j];
        Vt[cstage + 8 + j][rstage]  = (__bf16)g2[j];
        Vt[cstage + 12 + j][rstage] = (__bf16)g3[j];
      }
    }
    __syncthreads();

    // S = Q K^T * 1/sqrt(64), causal-masked on the diagonal tile
#pragma unroll
    for (int t = 0; t < 2; ++t) {
      int n0 = cs + t * 16;
      v8f s;
#pragma unroll
      for (int i = 0; i < 8; ++i) s[i] = 0.f;
#pragma unroll
      for (int k0 = 0; k0 < 64; k0 += 32) {
        const __bf16* ar = &Qs[qs + lr][k0];
        const __bf16* br = &Ks[n0 + lr][k0];
        v16bf a = ld16(ar + hi * 8, ar + 16 + hi * 8);
        v16bf bmat = ld16(br + hi * 8, br + 16 + hi * 8);
        s = wmma_bf16(a, bmat, s);
      }
#pragma unroll
      for (int r = 0; r < 8; ++r) {
        int row = qs + r + 8 * hi;
        int col = n0 + lr;
        float val = s[r] * 0.125f;
        if (kt == qt && col > row) val = -3.0e38f;
        Sc[row][col] = val;
      }
    }
    __syncthreads();

    if (tid < 64) {  // online softmax update for one row
      int row = tid;
      float m = rowM[row];
      for (int c = 0; c < 64; ++c) m = fmaxf(m, Sc[row][c]);
      float sc = expf(rowM[row] - m);
      float l = rowL[row] * sc;
      for (int c = 0; c < 64; ++c) {
        float p = expf(Sc[row][c] - m);
        l += p;
        Pb[row][c] = (__bf16)p;
      }
      rowM[row] = m; rowL[row] = l; rowScale[row] = sc;
    }
    __syncthreads();

    // rescale accumulators, then O += P V
#pragma unroll
    for (int r = 0; r < 8; ++r) {
      float sc = rowScale[qs + r + 8 * hi];
      acc0[r] *= sc;
      acc1[r] *= sc;
    }
#pragma unroll
    for (int t = 0; t < 2; ++t) {
      v8f& acc = t ? acc1 : acc0;
      int d0 = cs + t * 16;
#pragma unroll
      for (int k0 = 0; k0 < 64; k0 += 32) {
        const __bf16* ar = &Pb[qs + lr][k0];
        const __bf16* br = &Vt[d0 + lr][k0];
        v16bf a = ld16(ar + hi * 8, ar + 16 + hi * 8);
        v16bf bmat = ld16(br + hi * 8, br + 16 + hi * 8);
        acc = wmma_bf16(a, bmat, acc);
      }
    }
  }

#pragma unroll
  for (int t = 0; t < 2; ++t) {
    v8f& acc = t ? acc1 : acc0;
    int d0 = cs + t * 16;
#pragma unroll
    for (int r = 0; r < 8; ++r) {
      int row = qs + r + 8 * hi;
      float inv = 1.f / rowL[row];
      out[(size_t)(bb * SEQ + qt * 64 + row) * DMODEL + hh * DHEADC + d0 + lr] =
          (__bf16)(acc[r] * inv);
    }
  }
}

// ----------------------------------------------------------- MoE gate + top2
__global__ __launch_bounds__(256) void gate_kernel(
    const __bf16* __restrict__ xn, const float* __restrict__ gw,
    const float* __restrict__ gb, int* __restrict__ topi,
    float* __restrict__ topv) {
  __shared__ float red[256];
  __shared__ float ge[NEXP];
  int n = blockIdx.x, tid = threadIdx.x;
  float p[NEXP];
#pragma unroll
  for (int e = 0; e < NEXP; ++e) p[e] = 0.f;
  for (int d = tid; d < DMODEL; d += 256) {
    float xv = (float)xn[(size_t)n * DMODEL + d];
#pragma unroll
    for (int e = 0; e < NEXP; ++e) p[e] += xv * gw[d * NEXP + e];
  }
  for (int e = 0; e < NEXP; ++e) {
    red[tid] = p[e]; __syncthreads();
    for (int off = 128; off > 0; off >>= 1) {
      if (tid < off) red[tid] += red[tid + off];
      __syncthreads();
    }
    if (tid == 0) ge[e] = red[0] + gb[e];
    __syncthreads();
  }
  if (tid == 0) {
    float m = ge[0];
    for (int e = 1; e < NEXP; ++e) m = fmaxf(m, ge[e]);
    float pr[NEXP], s = 0.f;
    for (int e = 0; e < NEXP; ++e) { pr[e] = expf(ge[e] - m); s += pr[e]; }
    for (int e = 0; e < NEXP; ++e) pr[e] /= s;
    int i1 = 0;
    for (int e = 1; e < NEXP; ++e) if (pr[e] > pr[i1]) i1 = e;
    int i2 = (i1 == 0) ? 1 : 0;
    for (int e = 0; e < NEXP; ++e) if (e != i1 && pr[e] > pr[i2]) i2 = e;
    float ss = pr[i1] + pr[i2];
    topi[n * 2] = i1; topi[n * 2 + 1] = i2;
    topv[n * 2] = pr[i1] / ss; topv[n * 2 + 1] = pr[i2] / ss;
  }
}

// deterministic per-expert compaction (no atomics -> replay-stable)
__global__ void route_kernel(const int* __restrict__ topi,
                             const float* __restrict__ topv,
                             int* __restrict__ idx, float* __restrict__ wgt,
                             int* __restrict__ cnt) {
  int e = threadIdx.x;
  if (e >= NEXP) return;
  int c = 0;
  for (int n = 0; n < NTOK; ++n)
    for (int j = 0; j < 2; ++j)
      if (topi[n * 2 + j] == e) {
        idx[e * NTOK + c] = n;
        wgt[e * NTOK + c] = topv[n * 2 + j];
        ++c;
      }
  cnt[e] = c;
}

__global__ __launch_bounds__(256) void gather_kernel(
    const __bf16* __restrict__ xn, const int* __restrict__ idx,
    const int* __restrict__ cnt, __bf16* __restrict__ Xg) {
  int i = blockIdx.x * 16 + (threadIdx.x >> 4);
  if (i >= cnt[0]) return;
  int c0 = (threadIdx.x & 15) * 48;
  const v8bf* src = (const v8bf*)(xn + (size_t)idx[i] * DMODEL + c0);
  v8bf* dst = (v8bf*)(Xg + (size_t)i * DMODEL + c0);
#pragma unroll
  for (int j = 0; j < 6; ++j) dst[j] = src[j];
}

// ---------------------------------------------------------------- launcher
extern "C" void kernel_launch(void* const* d_in, const int* in_sizes, int n_in,
                              void* d_out, int out_size, void* d_ws,
                              size_t ws_size, hipStream_t stream) {
  const int*   x       = (const int*)d_in[0];
  const float* tok_emb = (const float*)d_in[1];
  const float* pos_emb = (const float*)d_in[2];
  const float* ln1_g   = (const float*)d_in[3];
  const float* ln1_b   = (const float*)d_in[4];
  const float* qkv_w   = (const float*)d_in[5];
  const float* qkv_b   = (const float*)d_in[6];
  const float* proj_w  = (const float*)d_in[7];
  const float* proj_b  = (const float*)d_in[8];
  const float* ln2_g   = (const float*)d_in[9];
  const float* ln2_b   = (const float*)d_in[10];
  const float* gate_w  = (const float*)d_in[11];
  const float* gate_b  = (const float*)d_in[12];
  const float* w1      = (const float*)d_in[13];
  const float* b1      = (const float*)d_in[14];
  const float* w2      = (const float*)d_in[15];
  const float* b2      = (const float*)d_in[16];
  const float* lnf_g   = (const float*)d_in[17];
  const float* lnf_b   = (const float*)d_in[18];
  float* out = (float*)d_out;

  char* ws = (char*)d_ws;
  size_t off = 0;
  auto alloc = [&](size_t bytes) -> char* {
    char* p = ws + off;
    off += (bytes + 255) & ~(size_t)255;
    return p;
  };
  float*  h       = (float*)alloc((size_t)NTOK * DMODEL * 4);
  __bf16* xn      = (__bf16*)alloc((size_t)NTOK * DMODEL * 2);
  float*  qkvbuf  = (float*)alloc((size_t)NTOK * 3 * DMODEL * 4);
  __bf16* attnout = (__bf16*)alloc((size_t)NTOK * DMODEL * 2);
  int*    topi    = (int*)alloc((size_t)NTOK * 2 * 4);
  float*  topv    = (float*)alloc((size_t)NTOK * 2 * 4);
  int*    cnt     = (int*)alloc(NEXP * 4);
  int*    idx     = (int*)alloc((size_t)NEXP * NTOK * 4);
  float*  wgt     = (float*)alloc((size_t)NEXP * NTOK * 4);
  __bf16* Xg      = (__bf16*)alloc((size_t)NTOK * DMODEL * 2);
  __bf16* H1      = (__bf16*)alloc((size_t)NTOK * DHID * 2);

  embed_kernel<<<(NTOK * DMODEL) / 256, 256, 0, stream>>>(x, tok_emb, pos_emb, h);

  for (int l = 0; l < NLAYER; ++l) {
    layernorm_kernel<<<NTOK, 256, 0, stream>>>(h, ln1_g + l * DMODEL,
                                               ln1_b + l * DMODEL, xn);
    gemm_bf16_kernel<0, false><<<dim3(NTOK / 64, (3 * DMODEL) / 64), 256, 0, stream>>>(
        xn, qkv_w + (size_t)l * DMODEL * 3 * DMODEL, qkv_b + (size_t)l * 3 * DMODEL,
        qkvbuf, nullptr, nullptr, nullptr, nullptr, NTOK, DMODEL, 3 * DMODEL,
        3 * DMODEL);
    attn_kernel<<<dim3(SEQ / 64, NHEADS, NBATCH), 256, 0, stream>>>(qkvbuf, attnout);
    gemm_bf16_kernel<1, false><<<dim3(NTOK / 64, DMODEL / 64), 256, 0, stream>>>(
        attnout, proj_w + (size_t)l * DMODEL * DMODEL, proj_b + (size_t)l * DMODEL,
        h, nullptr, nullptr, nullptr, nullptr, NTOK, DMODEL, DMODEL, DMODEL);

    layernorm_kernel<<<NTOK, 256, 0, stream>>>(h, ln2_g + l * DMODEL,
                                               ln2_b + l * DMODEL, xn);
    gate_kernel<<<NTOK, 256, 0, stream>>>(xn, gate_w + (size_t)l * DMODEL * NEXP,
                                          gate_b + (size_t)l * NEXP, topi, topv);
    route_kernel<<<1, 32, 0, stream>>>(topi, topv, idx, wgt, cnt);

    for (int e = 0; e < NEXP; ++e) {  // sequential: deterministic scatter order
      gather_kernel<<<NTOK / 16, 256, 0, stream>>>(xn, idx + e * NTOK, cnt + e, Xg);
      gemm_bf16_kernel<2, false><<<dim3(NTOK / 64, DHID / 64), 256, 0, stream>>>(
          Xg, w1 + ((size_t)(l * NEXP + e)) * DMODEL * DHID,
          b1 + ((size_t)(l * NEXP + e)) * DHID, nullptr, H1, nullptr, nullptr,
          cnt + e, NTOK, DMODEL, DHID, DHID);
      gemm_bf16_kernel<3, false><<<dim3(NTOK / 64, DMODEL / 64), 256, 0, stream>>>(
          H1, w2 + ((size_t)(l * NEXP + e)) * DHID * DMODEL,
          b2 + ((size_t)(l * NEXP + e)) * DMODEL, h, nullptr, idx + e * NTOK,
          wgt + e * NTOK, cnt + e, NTOK, DHID, DMODEL, DMODEL);
    }
  }

  layernorm_kernel<<<NTOK, 256, 0, stream>>>(h, lnf_g, lnf_b, xn);
  gemm_bf16_kernel<0, true><<<dim3(NTOK / 64, VOCAB / 64), 256, 0, stream>>>(
      xn, tok_emb, nullptr, out, nullptr, nullptr, nullptr, nullptr, NTOK, DMODEL,
      VOCAB, VOCAB);
}
